// GATv2Net_7086696038497
// MI455X (gfx1250) — matrix operand order repaired and checked by reading, not compiled, not run on hardware
//
#include <hip/hip_runtime.h>
#include <hip/hip_bf16.h>

// ---------------------------------------------------------------------------
// GATv2Net forward for MI455X (gfx1250, wave32, WMMA).
// Dense GEMMs -> v_wmma_f32_16x16x32_f16, 16x64 tile/wave (1 A frag : 4 WMMA),
//   A pre-converted to f16, W pre-packed into B-fragment order (b128 loads).
// Segment softmax -> ordered-uint atomicMax + f32 atomicAdd scatter.
// ---------------------------------------------------------------------------

#define NNODE  50000
#define NEDGE  1600000
#define NET    1650000      // E + N self loops
#define NGRAPH 512
#define HC_    128

typedef __attribute__((ext_vector_type(16))) _Float16 v16h;
typedef __attribute__((ext_vector_type(8)))  _Float16 v8h;
typedef __attribute__((ext_vector_type(8)))  float    v8f;

__device__ __forceinline__ unsigned fenc(float f) {
    unsigned u = __float_as_uint(f);
    return (u >> 31) ? ~u : (u | 0x80000000u);
}
__device__ __forceinline__ float fdec(unsigned e) {
    return (e >> 31) ? __uint_as_float(e & 0x7fffffffu) : __uint_as_float(~e);
}
#define ENC_NEG_INF 0x007FFFFFu   // fenc(-inf)

// ---------------------------------------------------------------------------
// f32 -> f16 row-major copy (8 elements/thread, b128 in / b128 out)
// ---------------------------------------------------------------------------
__global__ void k_cvt_f16(const float* __restrict__ in, _Float16* __restrict__ out, int n8) {
    int i = blockIdx.x * blockDim.x + threadIdx.x;
    if (i >= n8) return;
    const float4* p = (const float4*)(in + (long)i * 8);
    float4 a = p[0], b = p[1];
    v8h o;
    o[0] = (_Float16)a.x; o[1] = (_Float16)a.y; o[2] = (_Float16)a.z; o[3] = (_Float16)a.w;
    o[4] = (_Float16)b.x; o[5] = (_Float16)b.y; o[6] = (_Float16)b.z; o[7] = (_Float16)b.w;
    *(v8h*)(out + (long)i * 8) = o;
}

// ---------------------------------------------------------------------------
// Pack W[K,Nc] (f32 row-major) into WMMA B-fragment order (f16):
//   Wpk[((tn*kSteps + ks)*4 + s)*32 + lane][e] ,  e = 0..15
//   k = ks*32 + (lane>>4)*16 + e ,  n = tn*64 + s*16 + (lane&15)
// One thread per output element; writes are fully coalesced.
// ---------------------------------------------------------------------------
__global__ void k_pack_w(const float* __restrict__ W, _Float16* __restrict__ Wpk,
                         int K, int Nc) {
    int idx = blockIdx.x * blockDim.x + threadIdx.x;
    if (idx >= K * Nc) return;
    const int kSteps = K >> 5;
    int e    = idx & 15;
    int lane = (idx >> 4) & 31;
    int s    = (idx >> 9) & 3;
    int rest = idx >> 11;                 // tn*kSteps + ks
    int ks   = rest % kSteps;
    int tn   = rest / kSteps;
    int k = ks * 32 + (lane >> 4) * 16 + e;
    int n = tn * 64 + s * 16 + (lane & 15);
    Wpk[idx] = (_Float16)W[(long)k * Nc + n];
}

// ---------------------------------------------------------------------------
// Generic GEMM: C[M,Nc] = A16[M,K] @ W + bias  (f16 WMMA inner, f32 out).
// One wave computes a 16x64 tile: one A fragment drives 4 WMMAs per K-step.
// A16: row-major f16. Wpk: fragment-packed f16 (see k_pack_w).
// Preconditions: M % 16 == 0, Nc % 64 == 0, K % 32 == 0.
// ---------------------------------------------------------------------------
__global__ __launch_bounds__(128)
void k_gemm_wmma(const _Float16* __restrict__ A16, const _Float16* __restrict__ Wpk,
                 const float* __restrict__ bias, float* __restrict__ C,
                 int M, int K, int Nc) {
    const int lane = threadIdx.x & 31;
    const int wave = threadIdx.x >> 5;
    const int tilesN = Nc >> 6;                       // 64-wide N tiles
    const int kSteps = K >> 5;
    const long tiles = (long)(M >> 4) * tilesN;
    const long tile  = (long)blockIdx.x * (blockDim.x >> 5) + wave;
    if (tile >= tiles) return;                        // whole-wave exit (EXEC stays full)
    const int tm = (int)(tile / tilesN);
    const int tn = (int)(tile % tilesN);
    const int m0 = tm << 4, n0 = tn << 6;
    const int half = lane >> 4;                       // 0 | 1
    const int l16  = lane & 15;

    v8f acc[4];
#pragma unroll
    for (int s = 0; s < 4; ++s) {
        float b = bias ? bias[n0 + s * 16 + l16] : 0.0f;
#pragma unroll
        for (int r = 0; r < 8; ++r) acc[s][r] = b;
    }

    // A fragment lane row, contiguous halves (matches k_pack_w B layout)
    const _Float16* arow = A16 + (long)(m0 + l16) * K + half * 16;
    // B fragments for this lane, packed contiguously per (ks, s)
    const _Float16* bp = Wpk + ((long)tn * kSteps * 4 * 32 + lane) * 16;

    for (int ks = 0; ks < kSteps; ++ks) {
        if (ks + 1 < kSteps)                          // stream-ahead on A (global_prefetch_b8)
            __builtin_prefetch(arow + (ks + 1) * 32, 0, 3);
        v8h alo = *(const v8h*)(arow + ks * 32);      // K = ks*32 + half*16 + 0..7
        v8h ahi = *(const v8h*)(arow + ks * 32 + 8);  // K = ks*32 + half*16 + 8..15
        v16h af;
#pragma unroll
        for (int e = 0; e < 8; ++e) { af[e] = alo[e]; af[e + 8] = ahi[e]; }
        const _Float16* bks = bp + (long)ks * 4 * 32 * 16;
#pragma unroll
        for (int s = 0; s < 4; ++s) {
            v16h bf = *(const v16h*)(bks + (long)s * 32 * 16);
            acc[s] = __builtin_amdgcn_wmma_f32_16x16x32_f16(
                         false, af, false, bf, (short)0, acc[s], false, false);
        }
    }
    // D layout: element r of lane -> row m0 + r + 8*half, col n0 + s*16 + l16
#pragma unroll
    for (int s = 0; s < 4; ++s) {
        float* cp = C + (long)(m0 + half * 8) * Nc + n0 + s * 16 + l16;
#pragma unroll
        for (int r = 0; r < 8; ++r)
            cp[(long)r * Nc] = acc[s][r];
    }
}

// ---------------------------------------------------------------------------
__global__ void k_fill_u32(unsigned* p, unsigned v, int n) {
    int i = blockIdx.x * blockDim.x + threadIdx.x;
    if (i < n) p[i] = v;
}

// edge_attr column mean (partial sums -> 16 atomics per block)
#define EA_ROWS 8192
__global__ __launch_bounds__(256)
void k_ea_sum(const float* __restrict__ ea, int E, float* __restrict__ sum16) {
    const int k = threadIdx.x & 15;
    const int rl = threadIdx.x >> 4;                // 0..15
    long r0 = (long)blockIdx.x * EA_ROWS;
    long r1 = r0 + EA_ROWS; if (r1 > E) r1 = E;
    float acc = 0.f;
    for (long r = r0 + rl; r < r1; r += 16) acc += ea[r * 16 + k];
    __shared__ float s[256];
    s[threadIdx.x] = acc; __syncthreads();
    for (int st = 128; st >= 16; st >>= 1) {
        if (threadIdx.x < st) s[threadIdx.x] += s[threadIdx.x + st];
        __syncthreads();
    }
    if (threadIdx.x < 16) atomicAdd(&sum16[threadIdx.x], s[threadIdx.x]);
}
__global__ void k_ea_fin(const float* sum16, float* mean16, float invE) {
    if (threadIdx.x < 16) mean16[threadIdx.x] = sum16[threadIdx.x] * invE;
}

// ---------------------------------------------------------------------------
// Edge logits: one thread per (edge, head).
// logit = sum_c att[h,c] * lrelu(xl[src,h,c] + xr[dst,h,c] + (ea @ We)[h,c], 0.2)
// We cached in LDS as [c][h][k] (broadcast within h-group, 4 banks across heads).
// ---------------------------------------------------------------------------
__global__ __launch_bounds__(256)
void k_edge_logits(const float* __restrict__ xl, const float* __restrict__ xr,
                   const float* __restrict__ ea, const float* __restrict__ ea_mean,
                   const float* __restrict__ We, const float* __restrict__ att,
                   const int* __restrict__ ei,
                   float* __restrict__ logit, unsigned* __restrict__ mx) {
    __shared__ float sW[32 * 4 * 16];   // [c][h][k]
    __shared__ float sA[32 * 4];        // [c][h]
    for (int i = threadIdx.x; i < 2048; i += blockDim.x) {
        int c = i >> 6, h = (i >> 4) & 3, k = i & 15;
        sW[i] = We[k * 128 + h * 32 + c];
    }
    for (int i = threadIdx.x; i < 128; i += blockDim.x) {
        int c = i >> 2, h = i & 3;
        sA[i] = att[h * 32 + c];
    }
    __syncthreads();

    long t = (long)blockIdx.x * blockDim.x + threadIdx.x;
    if (t >= (long)NET * 4) return;
    const int h = (int)(t & 3);
    const long e = t >> 2;
    int src, dst;
    float eav[16];
    if (e < NEDGE) {
        src = ei[e]; dst = ei[NEDGE + e];
        const float4* p = (const float4*)(ea + e * 16);
#pragma unroll
        for (int q = 0; q < 4; ++q) { float4 v = p[q];
            eav[4*q] = v.x; eav[4*q+1] = v.y; eav[4*q+2] = v.z; eav[4*q+3] = v.w; }
    } else {
        src = dst = (int)(e - NEDGE);
#pragma unroll
        for (int k = 0; k < 16; ++k) eav[k] = ea_mean[k];
    }
    float xlv[32], xrv[32];
    {
        const float4* pl = (const float4*)(xl + (long)src * HC_ + h * 32);
        const float4* pr = (const float4*)(xr + (long)dst * HC_ + h * 32);
#pragma unroll
        for (int q = 0; q < 8; ++q) {
            float4 a = pl[q], b = pr[q];
            xlv[4*q] = a.x; xlv[4*q+1] = a.y; xlv[4*q+2] = a.z; xlv[4*q+3] = a.w;
            xrv[4*q] = b.x; xrv[4*q+1] = b.y; xrv[4*q+2] = b.z; xrv[4*q+3] = b.w;
        }
    }
    float acc = 0.f;
#pragma unroll
    for (int c = 0; c < 32; ++c) {
        const float4* wp = (const float4*)&sW[(c * 4 + h) * 16];
        float4 w0 = wp[0], w1 = wp[1], w2 = wp[2], w3 = wp[3];
        float eec = w0.x*eav[0]  + w0.y*eav[1]  + w0.z*eav[2]  + w0.w*eav[3]
                  + w1.x*eav[4]  + w1.y*eav[5]  + w1.z*eav[6]  + w1.w*eav[7]
                  + w2.x*eav[8]  + w2.y*eav[9]  + w2.z*eav[10] + w2.w*eav[11]
                  + w3.x*eav[12] + w3.y*eav[13] + w3.z*eav[14] + w3.w*eav[15];
        float v = xlv[c] + xrv[c] + eec;
        v = v > 0.f ? v : 0.2f * v;
        acc += sA[c * 4 + h] * v;
    }
    logit[t] = acc;
    atomicMax(&mx[(long)dst * 4 + h], fenc(acc));
}

// exp(logit - max), accumulate denominator
__global__ __launch_bounds__(256)
void k_edge_ez(const int* __restrict__ ei, float* __restrict__ logit,
               const unsigned* __restrict__ mx, float* __restrict__ den) {
    long t = (long)blockIdx.x * blockDim.x + threadIdx.x;
    if (t >= (long)NET * 4) return;
    const int h = (int)(t & 3);
    const long e = t >> 2;
    const int dst = (e < NEDGE) ? ei[NEDGE + e] : (int)(e - NEDGE);
    float z = __expf(logit[t] - fdec(mx[(long)dst * 4 + h]));
    logit[t] = z;
    atomicAdd(&den[(long)dst * 4 + h], z);
}

// out[dst] += alpha * xl[src]   (32 f32 atomics per (edge,head))
__global__ __launch_bounds__(256)
void k_edge_agg(const int* __restrict__ ei, const float* __restrict__ xl,
                const float* __restrict__ ez, const float* __restrict__ den,
                float* __restrict__ out) {
    long t = (long)blockIdx.x * blockDim.x + threadIdx.x;
    if (t >= (long)NET * 4) return;
    const int h = (int)(t & 3);
    const long e = t >> 2;
    int src, dst;
    if (e < NEDGE) { src = ei[e]; dst = ei[NEDGE + e]; }
    else           { src = dst = (int)(e - NEDGE); }
    const float alpha = ez[t] / den[(long)dst * 4 + h];
    const float4* xs = (const float4*)(xl + (long)src * HC_ + h * 32);
    float* od = out + (long)dst * HC_ + h * 32;
    float vals[32];
#pragma unroll
    for (int q = 0; q < 8; ++q) {
        float4 v = xs[q];
        vals[4*q] = v.x; vals[4*q+1] = v.y; vals[4*q+2] = v.z; vals[4*q+3] = v.w;
    }
#pragma unroll
    for (int c = 0; c < 32; ++c) atomicAdd(&od[c], alpha * vals[c]);
}

// ---------------------------------------------------------------------------
__global__ void k_bias_act(const float* __restrict__ in, const float* __restrict__ bias,
                           float* __restrict__ out, int n, int cols, float slope) {
    int i = blockIdx.x * blockDim.x + threadIdx.x;
    if (i >= n) return;
    float v = in[i] + (bias ? bias[i % cols] : 0.f);
    out[i] = v > 0.f ? v : slope * v;
}

#define STAT_ROWS 512
__global__ __launch_bounds__(128)
void k_colstats(const float* __restrict__ a, int rows, int cols,
                float* __restrict__ sum, float* __restrict__ sumsq) {
    int ch = blockIdx.y * 128 + threadIdx.x;
    if (ch >= cols) return;
    int r0 = blockIdx.x * STAT_ROWS;
    int r1 = r0 + STAT_ROWS; if (r1 > rows) r1 = rows;
    float s = 0.f, q = 0.f;
    for (int r = r0; r < r1; ++r) {
        float v = a[(long)r * cols + ch];
        s += v; q += v * v;
    }
    atomicAdd(&sum[ch], s);
    atomicAdd(&sumsq[ch], q);
}
__global__ void k_bn_finalize(const float* sum, const float* sumsq,
                              const float* g, const float* b, int cols,
                              float inv_rows, float* scale, float* shift) {
    int ch = blockIdx.x * blockDim.x + threadIdx.x;
    if (ch >= cols) return;
    float m = sum[ch] * inv_rows;
    float var = sumsq[ch] * inv_rows - m * m;
    float s = rsqrtf(var + 1e-5f) * g[ch];
    scale[ch] = s;
    shift[ch] = b[ch] - m * s;
}
__global__ void k_bn_apply(const float* __restrict__ in, const float* scale,
                           const float* shift, float* __restrict__ out,
                           int n, int cols) {
    int i = blockIdx.x * blockDim.x + threadIdx.x;
    if (i >= n) return;
    int c = i % cols;
    out[i] = in[i] * scale[c] + shift[c];
}

// ---------------------------------------------------------------------------
__global__ void k_pool(const float* __restrict__ v, const int* __restrict__ batch,
                       unsigned* __restrict__ gmax, float* __restrict__ gsum,
                       float* __restrict__ cnt) {
    int i = blockIdx.x * blockDim.x + threadIdx.x;
    if (i >= NNODE * HC_) return;
    int n = i >> 7, ch = i & 127;
    int g = batch[n];
    float val = v[i];
    atomicMax(&gmax[(long)g * HC_ + ch], fenc(val));
    atomicAdd(&gsum[(long)g * HC_ + ch], val);
    if (ch == 0) atomicAdd(&cnt[g], 1.0f);
}
__global__ void k_build_h(const unsigned* gmax, const float* gsum,
                          const float* cnt, float* h) {
    int i = blockIdx.x * blockDim.x + threadIdx.x;
    if (i >= NGRAPH * HC_) return;
    int g = i >> 7, ch = i & 127;
    h[(long)g * 256 + ch]        = fdec(gmax[i]);
    h[(long)g * 256 + 128 + ch]  = gsum[i] / cnt[g];
}
__global__ __launch_bounds__(256)
void k_final(const float* __restrict__ h3, const float* __restrict__ w,
             const float* __restrict__ b, float* __restrict__ out, int K) {
    int g = blockIdx.x;
    __shared__ float red[256];
    float acc = 0.f;
    for (int k = threadIdx.x; k < K; k += blockDim.x) acc += h3[(long)g * K + k] * w[k];
    red[threadIdx.x] = acc; __syncthreads();
    for (int s = 128; s > 0; s >>= 1) {
        if (threadIdx.x < s) red[threadIdx.x] += red[threadIdx.x + s];
        __syncthreads();
    }
    if (threadIdx.x == 0) out[g] = red[0] + b[0];
}

// ---------------------------------------------------------------------------
static inline void cvt16(const float* in, _Float16* out, long n, hipStream_t s) {
    int n8 = (int)(n / 8);
    k_cvt_f16<<<(n8 + 255) / 256, 256, 0, s>>>(in, out, n8);
}
// pack W then run GEMM (A16 already converted)
static inline void gemm(const _Float16* A16, const float* W, _Float16* wpk,
                        const float* b, float* C, int M, int K, int Nc, hipStream_t s) {
    int nw = K * Nc;
    k_pack_w<<<(nw + 255) / 256, 256, 0, s>>>(W, wpk, K, Nc);
    long tiles = (long)(M >> 4) * (Nc >> 6);          // 16x64 tiles per wave
    unsigned blocks = (unsigned)((tiles + 3) / 4);
    k_gemm_wmma<<<blocks, 128, 0, s>>>(A16, wpk, b, C, M, K, Nc);
}
static inline void bn_block(const float* in, float* out, const float* g, const float* b,
                            int rows, int cols, float* sum, float* sumsq,
                            float* scale, float* shift, hipStream_t s) {
    hipMemsetAsync(sum, 0, (size_t)cols * 4, s);
    hipMemsetAsync(sumsq, 0, (size_t)cols * 4, s);
    dim3 gs((rows + STAT_ROWS - 1) / STAT_ROWS, (cols + 127) / 128);
    k_colstats<<<gs, 128, 0, s>>>(in, rows, cols, sum, sumsq);
    k_bn_finalize<<<(cols + 127) / 128, 128, 0, s>>>(sum, sumsq, g, b, cols,
                                                     1.0f / (float)rows, scale, shift);
    int n = rows * cols;
    k_bn_apply<<<(n + 255) / 256, 256, 0, s>>>(in, scale, shift, out, n, cols);
}

extern "C" void kernel_launch(void* const* d_in, const int* in_sizes, int n_in,
                              void* d_out, int out_size, void* d_ws, size_t ws_size,
                              hipStream_t stream) {
    // ---- inputs: top-level insertion order, params pytree sorted-key flat ----
    const float* x         = (const float*)d_in[0];
    const float* edge_attr = (const float*)d_in[1];
    const float* b1    = (const float*)d_in[2];
    const float* b2    = (const float*)d_in[3];
    const float* b3    = (const float*)d_in[4];
    const float* fc1_b = (const float*)d_in[5];
    const float* fc1_w = (const float*)d_in[6];
    const float* fc2_b = (const float*)d_in[7];
    const float* fc2_w = (const float*)d_in[8];
    const float* fc3_b = (const float*)d_in[9];
    const float* fc3_w = (const float*)d_in[10];
    const float* g1    = (const float*)d_in[11];
    const float* g2    = (const float*)d_in[12];
    const float* g3    = (const float*)d_in[13];
    const float* l0_We = (const float*)d_in[14];
    const float* l0_Wl = (const float*)d_in[15];
    const float* l0_Wr = (const float*)d_in[16];
    const float* l0_at = (const float*)d_in[17];
    const float* l0_bl = (const float*)d_in[18];
    const float* l0_bo = (const float*)d_in[19];
    const float* l0_br = (const float*)d_in[20];
    const float* l0_bt = (const float*)d_in[21];
    const float* l0_g  = (const float*)d_in[22];
    const float* ly_We = (const float*)d_in[23];
    const float* ly_Wl = (const float*)d_in[24];
    const float* ly_Wr = (const float*)d_in[25];
    const float* ly_at = (const float*)d_in[26];
    const float* ly_bl = (const float*)d_in[27];
    const float* ly_bo = (const float*)d_in[28];
    const float* ly_br = (const float*)d_in[29];
    const float* ly_bt = (const float*)d_in[30];
    const float* ly_g  = (const float*)d_in[31];
    const float* out_b = (const float*)d_in[32];
    const float* out_w = (const float*)d_in[33];
    const int* edge_index = (const int*)d_in[34];
    const int* batch      = (const int*)d_in[35];

    // ---- workspace layout ----
    size_t off = 0;
    char* base = (char*)d_ws;
    auto alloc = [&](size_t bytes) -> char* {
        char* p = base + off;
        off = (off + bytes + 255) & ~(size_t)255;
        return p;
    };
    float*     xl     = (float*)alloc((size_t)NNODE * HC_ * 4);
    float*     xr     = (float*)alloc((size_t)NNODE * HC_ * 4);
    float*     vbuf   = (float*)alloc((size_t)NNODE * HC_ * 4);
    float*     outbuf = (float*)alloc((size_t)NNODE * HC_ * 4);
    _Float16*  a16    = (_Float16*)alloc((size_t)NNODE * 256 * 2);  // largest A (layer 0)
    _Float16*  wpk    = (_Float16*)alloc((size_t)1024 * 1024 * 2);  // largest W (fc2)
    float*     ezbuf  = (float*)alloc((size_t)NET * 4 * 4);
    unsigned*  mx     = (unsigned*)alloc((size_t)NNODE * 4 * 4);
    float*     den    = (float*)alloc((size_t)NNODE * 4 * 4);
    float*     ssum   = (float*)alloc(1024 * 4);
    float*     ssq    = (float*)alloc(1024 * 4);
    float*     sscale = (float*)alloc(1024 * 4);
    float*     sshift = (float*)alloc(1024 * 4);
    float*     easum  = (float*)alloc(16 * 4);
    float*     eamean = (float*)alloc(16 * 4);
    unsigned*  gmax   = (unsigned*)alloc((size_t)NGRAPH * HC_ * 4);
    float*     gsum   = (float*)alloc((size_t)NGRAPH * HC_ * 4);
    float*     cnt    = (float*)alloc((size_t)NGRAPH * 4);
    float*     hbuf   = (float*)alloc((size_t)NGRAPH * 256 * 4);
    float*     m1     = (float*)alloc((size_t)NGRAPH * 1024 * 4);
    float*     m2     = (float*)alloc((size_t)NGRAPH * 512 * 4);
    float*     m3     = (float*)alloc((size_t)NGRAPH * 256 * 4);
    if (off > ws_size) return;   // insufficient scratch: bail deterministically

    const long et4 = (long)NET * 4;
    const unsigned eblk = (unsigned)((et4 + 255) / 256);

    // ---- edge_attr mean (for self-loop fill) ----
    hipMemsetAsync(easum, 0, 16 * 4, stream);
    k_ea_sum<<<(NEDGE + EA_ROWS - 1) / EA_ROWS, 256, 0, stream>>>(edge_attr, NEDGE, easum);
    k_ea_fin<<<1, 32, 0, stream>>>(easum, eamean, 1.0f / (float)NEDGE);

    // ---- 5 GATv2 layers ----
    for (int layer = 0; layer < 5; ++layer) {
        const float *Wl, *bl, *Wr, *br, *We, *at, *bo, *gg, *bt, *Ain;
        int K;
        if (layer == 0) {
            Ain = x; K = 256;
            Wl = l0_Wl; bl = l0_bl; Wr = l0_Wr; br = l0_br;
            We = l0_We; at = l0_at; bo = l0_bo; gg = l0_g; bt = l0_bt;
        } else {
            int i = layer - 1;
            Ain = vbuf; K = 128;
            Wl = ly_Wl + (size_t)i * 128 * 128; bl = ly_bl + (size_t)i * 128;
            Wr = ly_Wr + (size_t)i * 128 * 128; br = ly_br + (size_t)i * 128;
            We = ly_We + (size_t)i * 16 * 128;  at = ly_at + (size_t)i * 128;
            bo = ly_bo + (size_t)i * 128; gg = ly_g + (size_t)i * 128;
            bt = ly_bt + (size_t)i * 128;
        }
        cvt16(Ain, a16, (long)NNODE * K, stream);              // A -> f16 once per layer
        gemm(a16, Wl, wpk, bl, xl, NNODE, K, HC_, stream);     // xl = v@Wl + bl (WMMA)
        gemm(a16, Wr, wpk, br, xr, NNODE, K, HC_, stream);     // xr = v@Wr + br (WMMA)

        hipMemsetAsync(outbuf, 0, (size_t)NNODE * HC_ * 4, stream);
        hipMemsetAsync(den, 0, (size_t)NNODE * 4 * 4, stream);
        k_fill_u32<<<(NNODE * 4 + 255) / 256, 256, 0, stream>>>(mx, ENC_NEG_INF, NNODE * 4);

        k_edge_logits<<<eblk, 256, 0, stream>>>(xl, xr, edge_attr, eamean, We, at,
                                                edge_index, ezbuf, mx);
        k_edge_ez<<<eblk, 256, 0, stream>>>(edge_index, ezbuf, mx, den);
        k_edge_agg<<<eblk, 256, 0, stream>>>(edge_index, xl, ezbuf, den, outbuf);

        int nelem = NNODE * HC_;
        k_bias_act<<<(nelem + 255) / 256, 256, 0, stream>>>(outbuf, bo, outbuf,
                                                            nelem, HC_, 0.01f);
        bn_block(outbuf, vbuf, gg, bt, NNODE, HC_, ssum, ssq, sscale, sshift, stream);
    }

    // ---- global max + mean pooling ----
    k_fill_u32<<<(NGRAPH * HC_ + 255) / 256, 256, 0, stream>>>(gmax, ENC_NEG_INF, NGRAPH * HC_);
    hipMemsetAsync(gsum, 0, (size_t)NGRAPH * HC_ * 4, stream);
    hipMemsetAsync(cnt, 0, (size_t)NGRAPH * 4, stream);
    k_pool<<<(NNODE * HC_ + 255) / 256, 256, 0, stream>>>(vbuf, batch, gmax, gsum, cnt);
    k_build_h<<<(NGRAPH * HC_ + 255) / 256, 256, 0, stream>>>(gmax, gsum, cnt, hbuf);

    // ---- MLP head (WMMA GEMMs) ----
    cvt16(hbuf, a16, (long)NGRAPH * 256, stream);
    gemm(a16, fc1_w, wpk, fc1_b, m1, NGRAPH, 256, 1024, stream);
    k_bias_act<<<(NGRAPH * 1024 + 255) / 256, 256, 0, stream>>>(m1, nullptr, m1,
                                                                NGRAPH * 1024, 1024, 0.01f);
    bn_block(m1, m1, g1, b1, NGRAPH, 1024, ssum, ssq, sscale, sshift, stream);

    cvt16(m1, a16, (long)NGRAPH * 1024, stream);
    gemm(a16, fc2_w, wpk, fc2_b, m2, NGRAPH, 1024, 512, stream);
    k_bias_act<<<(NGRAPH * 512 + 255) / 256, 256, 0, stream>>>(m2, nullptr, m2,
                                                               NGRAPH * 512, 512, 0.01f);
    bn_block(m2, m2, g2, b2, NGRAPH, 512, ssum, ssq, sscale, sshift, stream);

    cvt16(m2, a16, (long)NGRAPH * 512, stream);
    gemm(a16, fc3_w, wpk, fc3_b, m3, NGRAPH, 512, 256, stream);
    k_bias_act<<<(NGRAPH * 256 + 255) / 256, 256, 0, stream>>>(m3, nullptr, m3,
                                                               NGRAPH * 256, 256, 0.01f);
    bn_block(m3, m3, g3, b3, NGRAPH, 256, ssum, ssq, sscale, sshift, stream);

    k_final<<<NGRAPH, 256, 0, stream>>>(m3, out_w, out_b, (float*)d_out, 256);
}
